// AdaptiveTransformerModel_43628277792945
// MI455X (gfx1250) — matrix operand order repaired and checked by reading, not compiled, not run on hardware
//
#include <hip/hip_runtime.h>
#include <math.h>

// ---------------------------------------------------------------------------
// Model constants (match reference)
// ---------------------------------------------------------------------------
#define Dm   1024
#define Hh   16
#define HDm  64
#define Lm   8
#define Fm   4096
#define Vm   32000
#define Bm   4
#define Tm   512
#define BT   (Bm * Tm)          // 2048 tokens

// ---------------------------------------------------------------------------
// Vector types (CDNA5 / gfx1250, wave32)
// ---------------------------------------------------------------------------
typedef __attribute__((ext_vector_type(16))) __bf16 v16bf;
typedef __attribute__((ext_vector_type(8)))  float  v8f;
typedef __attribute__((ext_vector_type(4)))  float  v4f;
typedef __attribute__((ext_vector_type(4)))  int    v4i;

#define AS1 __attribute__((address_space(1)))
#define AS3 __attribute__((address_space(3)))

// gfx1250 async global->LDS copy (ASYNCcnt path). Guarded so the file still
// compiles (with a plain LDS-store fallback) if this toolchain lacks them.
#if defined(__HIP_DEVICE_COMPILE__) && \
    __has_builtin(__builtin_amdgcn_global_load_async_to_lds_b128) && \
    __has_builtin(__builtin_amdgcn_s_wait_asynccnt)
#define HAVE_ASYNC_LDS 1
#else
#define HAVE_ASYNC_LDS 0
#endif

// ---------------------------------------------------------------------------
// Fragment loaders using contiguous k order (same for A and B; ISA table order differs).
//
// 16-bit A-matrix 16x32 (MxK):
//   lanes 0-15 : row M = lane,   elems 0..15 = K k0+0..15
//   lanes16-31 : row M = lane-16,elems 0..15 = K k0+16..31
// ---------------------------------------------------------------------------
__device__ __forceinline__ v16bf load_a_frag(const float* __restrict__ A, int lda,
                                             int m0, int k0, int lane) {
  const int r  = m0 + (lane & 15);
  const int kb = k0 + ((lane >> 4) << 4);      // +0 or +16
  const float* p = A + (size_t)r * lda;
  v16bf a;
#pragma unroll
  for (int i = 0; i < 8; ++i)  a[i]     = (__bf16)p[kb + i];
#pragma unroll
  for (int i = 0; i < 8; ++i)  a[8 + i] = (__bf16)p[kb + 8 + i];
  return a;
}

// 16-bit B-matrix 32x16 (KxN):
//   lane L holds column N = n0 + (L&15); K = (L>=16 ? 16 : 0) + 0..15
// Source: weight matrix W[N][K] row-major (K contiguous) -> B = W^T.
__device__ __forceinline__ v16bf load_b_frag(const float* __restrict__ W, int ldb,
                                             int n0, int k0, int lane) {
  const int n  = n0 + (lane & 15);
  const int kb = k0 + ((lane >> 4) << 4);      // +0 or +16
  const float* p = W + (size_t)n * ldb + kb;
  v16bf b;
#pragma unroll
  for (int i = 0; i < 16; ++i) b[i] = (__bf16)p[i];
  return b;
}

// B fragment from the LDS-staged tile (row stride 36 floats, 16B aligned).
__device__ __forceinline__ v16bf load_b_frag_lds(const float* bstage,
                                                 int n0, int lane) {
  const int n  = n0 + (lane & 15);
  const int kb = (lane >> 4) << 4;
  const float* p = bstage + n * 36 + kb;
  v16bf b;
#pragma unroll
  for (int i = 0; i < 16; ++i) b[i] = (__bf16)p[i];
  return b;
}

// B fragment, source stored K-major: B[k][n] = V[k*ldv + n]  (for P·V).
__device__ __forceinline__ v16bf load_b_frag_kmajor(const float* __restrict__ V, int ldv,
                                                    int k0, int n0, int lane) {
  const int n  = n0 + (lane & 15);
  const int kb = k0 + ((lane >> 4) << 4);
  v16bf b;
#pragma unroll
  for (int i = 0; i < 16; ++i) b[i] = (__bf16)V[(size_t)(kb + i) * ldv + n];
  return b;
}

// C/D 16x16 f32 layout: VGPR g -> row m0+g (lanes 0-15) / m0+8+g (lanes 16-31),
// column n0 + (lane&15).
__device__ __forceinline__ void store_frag(float* __restrict__ C, int ldc,
                                           const float* __restrict__ bias,
                                           const float* __restrict__ resid,
                                           int m0, int n0, int lane,
                                           v8f c, int gelu) {
  const int n     = n0 + (lane & 15);
  const int mbase = m0 + ((lane >> 4) << 3);
  const float bv  = bias ? bias[n] : 0.0f;
#pragma unroll
  for (int g = 0; g < 8; ++g) {
    float val = c[g] + bv;
    if (gelu)  val = 0.5f * val * (1.0f + erff(val * 0.70710678118654752f));
    if (resid) val += resid[(size_t)(mbase + g) * ldc + n];
    C[(size_t)(mbase + g) * ldc + n] = val;
  }
}

// ---------------------------------------------------------------------------
// WMMA GEMM:  C[M,N] = A[M,K] * W[N,K]^T  (+bias) (+GELU) (+residual)
//
// Workgroup = 8 waves arranged 4(M) x 2(N); each wave owns a 32x32 register
// tile (2x2 fragments, fp32 accum). Workgroup output tile = 128 x 64.
// Per K-step of 32 the 64x32 B-tile is staged to LDS once (async global->LDS
// on gfx1250), then every wave issues 4 v_wmma_f32_16x16x32_bf16.
// Requires M % 128 == 0, N % 64 == 0, K % 32 == 0 (true for all GEMMs here).
// ---------------------------------------------------------------------------
__global__ void __launch_bounds__(256)
gemm_bf16_wmma_kernel(const float* __restrict__ A, int lda,
                      const float* __restrict__ W, int ldb,
                      const float* __restrict__ bias,
                      const float* __restrict__ resid,
                      float* __restrict__ C, int ldc,
                      int K, int gelu) {
  __shared__ float bstage[64 * 36];            // 9.2 KB of 320 KB WGP LDS

  const int wave = threadIdx.x >> 5;
  const int lane = threadIdx.x & 31;
  const int mWG  = blockIdx.x * 128;
  const int nWG  = blockIdx.y * 64;
  const int m0   = mWG + (wave & 3) * 32;      // wave's 32x32 tile origin
  const int nl0  = (wave >> 2) * 32;           // tile-local N origin (0 or 32)
  const int n0   = nWG + nl0;

  const int srow = threadIdx.x >> 2;           // 0..63 : B-tile row to stage
  const int sq   = threadIdx.x & 3;            // 0..3  : 8-float chunk

  v8f c00 = {}, c01 = {}, c10 = {}, c11 = {};

  for (int kk = 0; kk < K; kk += 32) {
    // ---- stage B tile [64 rows x 32 K] fp32 into LDS ----
    const float* src = W + (size_t)(nWG + srow) * ldb + kk + sq * 8;
    float*       dst = &bstage[srow * 36 + sq * 8];
#if HAVE_ASYNC_LDS
    __builtin_amdgcn_global_load_async_to_lds_b128((AS1 v4i*)src,
                                                   (AS3 v4i*)dst, 0, 0);
    __builtin_amdgcn_global_load_async_to_lds_b128((AS1 v4i*)(src + 4),
                                                   (AS3 v4i*)(dst + 4), 0, 0);
    __builtin_amdgcn_s_wait_asynccnt(0);
#else
    *(v4f*)dst       = *(const v4f*)src;
    *(v4f*)(dst + 4) = *(const v4f*)(src + 4);
#endif
    __syncthreads();

    // prefetch next A chunk (global_prefetch_b8)
    __builtin_prefetch(A + (size_t)(m0 + (lane & 15)) * lda + kk + 32, 0, 1);

    v16bf a0 = load_a_frag(A, lda, m0,      kk, lane);
    v16bf a1 = load_a_frag(A, lda, m0 + 16, kk, lane);
    v16bf b0 = load_b_frag_lds(bstage, nl0,      lane);
    v16bf b1 = load_b_frag_lds(bstage, nl0 + 16, lane);

    c00 = __builtin_amdgcn_wmma_f32_16x16x32_bf16(false, a0, false, b0,
                                                  (short)0, c00, false, false);
    c01 = __builtin_amdgcn_wmma_f32_16x16x32_bf16(false, a0, false, b1,
                                                  (short)0, c01, false, false);
    c10 = __builtin_amdgcn_wmma_f32_16x16x32_bf16(false, a1, false, b0,
                                                  (short)0, c10, false, false);
    c11 = __builtin_amdgcn_wmma_f32_16x16x32_bf16(false, a1, false, b1,
                                                  (short)0, c11, false, false);
    __syncthreads();                           // protect bstage for next step
  }

  store_frag(C, ldc, bias, resid, m0,      n0,      lane, c00, gelu);
  store_frag(C, ldc, bias, resid, m0,      n0 + 16, lane, c01, gelu);
  store_frag(C, ldc, bias, resid, m0 + 16, n0,      lane, c10, gelu);
  store_frag(C, ldc, bias, resid, m0 + 16, n0 + 16, lane, c11, gelu);
}

// ---------------------------------------------------------------------------
// Fused causal attention. One wave32 per (batch, head, 16-query tile):
//   scores[16,512] = (Q K^T)/8 via WMMA -> LDS; mask+softmax; O = P V via WMMA
// q/k/v/o are [B, T, H, HD] fp32 (row stride D between tokens).
// ---------------------------------------------------------------------------
__global__ void __launch_bounds__(32)
attn_wmma_kernel(const float* __restrict__ q, const float* __restrict__ k,
                 const float* __restrict__ v, float* __restrict__ o) {
  __shared__ float sc[16 * Tm];                // 32 KB
  const int lane = threadIdx.x;
  const int qt = blockIdx.x & 31;              // T/16 = 32
  const int hh = (blockIdx.x >> 5) & 15;
  const int b  = blockIdx.x >> 9;
  const int t0 = qt << 4;

  const float* Qb = q + (size_t)b * Tm * Dm + hh * HDm;
  const float* Kb = k + (size_t)b * Tm * Dm + hh * HDm;
  const float* Vb = v + (size_t)b * Tm * Dm + hh * HDm;

  const int nl = lane & 15;
  const int mb = (lane >> 4) << 3;

  // ---- scores = Q K^T * (1/sqrt(64)), causal-masked, into LDS ----
  for (int st = 0; st < Tm / 16; ++st) {
    v8f c = {};
#pragma unroll
    for (int kk = 0; kk < HDm; kk += 32) {
      v16bf a  = load_a_frag(Qb, Dm, t0, kk, lane);
      v16bf bb = load_b_frag(Kb, Dm, st * 16, kk, lane);
      c = __builtin_amdgcn_wmma_f32_16x16x32_bf16(false, a, false, bb,
                                                  (short)0, c, false, false);
    }
    const int s = st * 16 + nl;
#pragma unroll
    for (int g = 0; g < 8; ++g) {
      const int m = mb + g;
      float val = c[g] * 0.125f;
      if (s > t0 + m) val = -1.0e9f;           // causal mask
      sc[m * Tm + s] = val;
    }
  }
  __syncthreads();

  // ---- row softmax over 512 keys (wave32 shuffle reductions) ----
  for (int r = 0; r < 16; ++r) {
    float mx = -3.0e38f;
    for (int j = lane; j < Tm; j += 32) mx = fmaxf(mx, sc[r * Tm + j]);
#pragma unroll
    for (int off = 16; off; off >>= 1) mx = fmaxf(mx, __shfl_xor(mx, off, 32));
    float sum = 0.0f;
    for (int j = lane; j < Tm; j += 32) {
      float e = __expf(sc[r * Tm + j] - mx);
      sc[r * Tm + j] = e;
      sum += e;
    }
#pragma unroll
    for (int off = 16; off; off >>= 1) sum += __shfl_xor(sum, off, 32);
    const float inv = 1.0f / sum;
    for (int j = lane; j < Tm; j += 32) sc[r * Tm + j] *= inv;
  }
  __syncthreads();

  // ---- O = P V ----
  float* op = o + (size_t)b * Tm * Dm + hh * HDm;
  for (int nt = 0; nt < HDm / 16; ++nt) {
    v8f c = {};
    for (int kk = 0; kk < Tm; kk += 32) {
      v16bf a  = load_a_frag(sc, Tm, 0, kk, lane);          // P from LDS
      v16bf bb = load_b_frag_kmajor(Vb, Dm, kk, nt * 16, lane);
      c = __builtin_amdgcn_wmma_f32_16x16x32_bf16(false, a, false, bb,
                                                  (short)0, c, false, false);
    }
#pragma unroll
    for (int g = 0; g < 8; ++g)
      op[(size_t)(t0 + mb + g) * Dm + nt * 16 + nl] = c[g];
  }
}

// ---------------------------------------------------------------------------
// Elementwise helpers
// ---------------------------------------------------------------------------
__global__ void embed_kernel(const int* __restrict__ ids,
                             const float* __restrict__ wte,
                             const float* __restrict__ wpe,
                             float* __restrict__ h) {
  int i = blockIdx.x * blockDim.x + threadIdx.x;
  if (i >= BT * Dm) return;
  int d = i % Dm;
  int t = (i / Dm) % Tm;
  int b = i / (Dm * Tm);
  int id = ids[b * Tm + t];
  h[i] = wte[(size_t)id * Dm + d] + wpe[(size_t)t * Dm + d];
}

__global__ void __launch_bounds__(256)
ln_kernel(const float* __restrict__ in, const float* __restrict__ g,
          const float* __restrict__ bta, float* __restrict__ out) {
  __shared__ float red[256];
  const int row = blockIdx.x;
  const int tid = threadIdx.x;
  const float* p = in + (size_t)row * Dm;
  float s = 0.0f;
  for (int j = tid; j < Dm; j += 256) s += p[j];
  red[tid] = s; __syncthreads();
  for (int st = 128; st > 0; st >>= 1) { if (tid < st) red[tid] += red[tid + st]; __syncthreads(); }
  const float mean = red[0] * (1.0f / Dm);
  __syncthreads();
  float vv = 0.0f;
  for (int j = tid; j < Dm; j += 256) { float d = p[j] - mean; vv += d * d; }
  red[tid] = vv; __syncthreads();
  for (int st = 128; st > 0; st >>= 1) { if (tid < st) red[tid] += red[tid + st]; __syncthreads(); }
  const float inv = rsqrtf(red[0] * (1.0f / Dm) + 1e-5f);
  __syncthreads();
  float* q = out + (size_t)row * Dm;
  for (int j = tid; j < Dm; j += 256) q[j] = (p[j] - mean) * inv * g[j] + bta[j];
}

// Repack per-head output projection Wo[H,D,HD] (scaled by gate) into
// Wp[D, H*HD] with K-contiguous rows for the GEMM B-loader.
__global__ void pack_wo_kernel(const float* __restrict__ wo,
                               const float* __restrict__ gate,
                               float* __restrict__ wp) {
  int i = blockIdx.x * blockDim.x + threadIdx.x;
  if (i >= Dm * Dm) return;
  int d  = i / Dm;
  int j  = i % Dm;
  int hh = j / HDm;
  int hd = j % HDm;
  wp[i] = gate[hh] * wo[((size_t)hh * Dm + d) * HDm + hd];
}

__global__ void bosum_kernel(const float* __restrict__ bo,
                             const float* __restrict__ gate,
                             float* __restrict__ out) {
  int d = blockIdx.x * blockDim.x + threadIdx.x;
  if (d >= Dm) return;
  float s = 0.0f;
  for (int hh = 0; hh < Hh; ++hh) s += gate[hh] * bo[(size_t)hh * Dm + d];
  out[d] = s;
}

__global__ void concat_kernel(const float* __restrict__ a,
                              const float* __restrict__ b,
                              float* __restrict__ outp) {
  int i = blockIdx.x * blockDim.x + threadIdx.x;
  if (i >= BT * 2 * Dm) return;
  int m = i / (2 * Dm);
  int j = i % (2 * Dm);
  outp[i] = (j < Dm) ? a[(size_t)m * Dm + j] : b[(size_t)m * Dm + (j - Dm)];
}

// ---------------------------------------------------------------------------
// Host-side launch helpers
// ---------------------------------------------------------------------------
static void launch_gemm(const float* A, int lda, const float* W, int ldb,
                        const float* bias, const float* resid,
                        float* C, int ldc, int M, int N, int K, int gelu,
                        hipStream_t stream) {
  dim3 grid(M / 128, N / 64);
  gemm_bf16_wmma_kernel<<<grid, 256, 0, stream>>>(A, lda, W, ldb, bias, resid,
                                                  C, ldc, K, gelu);
}

extern "C" void kernel_launch(void* const* d_in, const int* in_sizes, int n_in,
                              void* d_out, int out_size, void* d_ws, size_t ws_size,
                              hipStream_t stream) {
  (void)in_sizes; (void)n_in; (void)out_size; (void)ws_size;

  const int*   ids  = (const int*)  d_in[0];
  const float* wte  = (const float*)d_in[1];
  const float* wpe  = (const float*)d_in[2];
  const float* Wq   = (const float*)d_in[3];
  const float* bq   = (const float*)d_in[4];
  const float* Wk   = (const float*)d_in[5];
  const float* bk   = (const float*)d_in[6];
  const float* Wv   = (const float*)d_in[7];
  const float* bv   = (const float*)d_in[8];
  const float* Wo   = (const float*)d_in[9];
  const float* bo   = (const float*)d_in[10];
  const float* gate = (const float*)d_in[11];
  const float* ln1g = (const float*)d_in[12];
  const float* ln1b = (const float*)d_in[13];
  const float* ln2g = (const float*)d_in[14];
  const float* ln2b = (const float*)d_in[15];
  const float* w1   = (const float*)d_in[16];
  const float* b1   = (const float*)d_in[17];
  const float* w2   = (const float*)d_in[18];
  const float* b2   = (const float*)d_in[19];
  const float* skw  = (const float*)d_in[20];
  const float* skb  = (const float*)d_in[21];
  const float* lnfg = (const float*)d_in[22];
  const float* lnfb = (const float*)d_in[23];

  // Workspace carve-out (~132 MB of fp32 scratch)
  float* ws   = (float*)d_ws;
  float* hb   = ws;  ws += (size_t)BT * Dm;        // hidden state
  float* xb   = ws;  ws += (size_t)BT * Dm;        // layernorm output
  float* qb   = ws;  ws += (size_t)BT * Dm;
  float* kb   = ws;  ws += (size_t)BT * Dm;
  float* vb   = ws;  ws += (size_t)BT * Dm;
  float* ob   = ws;  ws += (size_t)BT * Dm;        // attention output
  float* wp   = ws;  ws += (size_t)Dm * Dm;        // packed Wo
  float* bsum = ws;  ws += (size_t)Dm;             // combined bo
  float* ffb  = ws;  ws += (size_t)BT * Fm;        // MLP hidden
  float* fus  = ws;  ws += (size_t)BT * 2 * Dm;    // skip concat
  float* encb = ws;                                 // 4 encoder snapshots

  // Embedding: h = wte[ids] + wpe
  embed_kernel<<<(BT * Dm + 255) / 256, 256, 0, stream>>>(ids, wte, wpe, hb);

  for (int i = 0; i < Lm; ++i) {
    // --- attention block ---
    ln_kernel<<<BT, 256, 0, stream>>>(hb, ln1g + (size_t)i * Dm, ln1b + (size_t)i * Dm, xb);
    launch_gemm(xb, Dm, Wq + (size_t)i * Dm * Dm, Dm, bq + (size_t)i * Dm, nullptr,
                qb, Dm, BT, Dm, Dm, 0, stream);
    launch_gemm(xb, Dm, Wk + (size_t)i * Dm * Dm, Dm, bk + (size_t)i * Dm, nullptr,
                kb, Dm, BT, Dm, Dm, 0, stream);
    launch_gemm(xb, Dm, Wv + (size_t)i * Dm * Dm, Dm, bv + (size_t)i * Dm, nullptr,
                vb, Dm, BT, Dm, Dm, 0, stream);
    attn_wmma_kernel<<<Bm * Hh * (Tm / 16), 32, 0, stream>>>(qb, kb, vb, ob);

    pack_wo_kernel<<<(Dm * Dm + 255) / 256, 256, 0, stream>>>(
        Wo + (size_t)i * Hh * Dm * HDm, gate + (size_t)i * Hh, wp);
    bosum_kernel<<<(Dm + 255) / 256, 256, 0, stream>>>(
        bo + (size_t)i * Hh * Dm, gate + (size_t)i * Hh, bsum);
    // h = h + o @ Wp^T + bsum  (residual fused in epilogue)
    launch_gemm(ob, Dm, wp, Dm, bsum, hb, hb, Dm, BT, Dm, Dm, 0, stream);

    if (i < Lm / 2)
      (void)hipMemcpyAsync(encb + (size_t)i * BT * Dm, hb,
                           sizeof(float) * (size_t)BT * Dm,
                           hipMemcpyDeviceToDevice, stream);

    // --- MLP block ---
    ln_kernel<<<BT, 256, 0, stream>>>(hb, ln2g + (size_t)i * Dm, ln2b + (size_t)i * Dm, xb);
    launch_gemm(xb, Dm, w1 + (size_t)i * Fm * Dm, Dm, b1 + (size_t)i * Fm, nullptr,
                ffb, Fm, BT, Fm, Dm, 1 /*gelu*/, stream);
    launch_gemm(ffb, Fm, w2 + (size_t)i * Dm * Fm, Fm, b2 + (size_t)i * Dm, hb,
                hb, Dm, BT, Dm, Fm, 0, stream);

    // --- U-Net style skip fusion for decoder half ---
    if (i >= Lm / 2) {
      int el = Lm - i - 1;   // 3,2,1,0 -> always captured
      concat_kernel<<<(BT * 2 * Dm + 255) / 256, 256, 0, stream>>>(
          hb, encb + (size_t)el * BT * Dm, fus);
      launch_gemm(fus, 2 * Dm, skw + (size_t)i * Dm * 2 * Dm, 2 * Dm,
                  skb + (size_t)i * Dm, nullptr, hb, Dm, BT, Dm, 2 * Dm, 0, stream);
    }
  }

  // Final LN + tied lm_head: logits = ln(h) @ wte^T
  ln_kernel<<<BT, 256, 0, stream>>>(hb, lnfg, lnfb, xb);
  launch_gemm(xb, Dm, wte, Dm, nullptr, nullptr, (float*)d_out, Vm,
              BT, Vm, Dm, 0, stream);
}
